// Linear8Bit_47296179864030
// MI455X (gfx1250) — matrix operand order repaired and checked by reading, not compiled, not run on hardware
//
#include <hip/hip_runtime.h>
#include <hip/hip_bf16.h>
#include <cstdint>

typedef __attribute__((ext_vector_type(16))) _Float16 v16h;
typedef __attribute__((ext_vector_type(8)))  float    v8f;

// Exact operand types for __builtin_amdgcn_global_load_async_to_lds_b128:
// (v4i addrspace(1)* src, v4i addrspace(3)* dst, imm offset, imm cpol)
typedef int vs4i __attribute__((vector_size(16)));
typedef __attribute__((address_space(1))) vs4i gvs4i;   // global v4i
typedef __attribute__((address_space(3))) vs4i lvs4i;   // LDS v4i

#if __has_builtin(__builtin_amdgcn_global_load_async_to_lds_b128) && \
    __has_builtin(__builtin_amdgcn_s_wait_asynccnt)
#define USE_ASYNC 1
#else
#define USE_ASYNC 0
#endif

#define M_DIM 64
#define K_DIM 8192
#define N_DIM 8192
#define QB    32
#define NBLK  (K_DIM / QB)        // 256 quant blocks == WMMA K-steps
#define KSPLIT 4
#define KT_PER_WG (NBLK / KSPLIT) // 64 blocks per workgroup
#define NT    128                 // output columns per workgroup
#define NTILES (NT / 16)          // 8 WMMA column tiles per workgroup
#define ROW_I 128                 // ints per LDS row (one K-row of the tile)
// K-rows 16..31 live in a second region skewed by 16 ints (2064 % 64 == 16).
#define HI_OFF (16 * ROW_I + 16)  // 2064 ints
#define BUF_I  (HI_OFF + 16 * ROW_I) // 4112 ints = 16448 B per buffer

// ---------------------------------------------------------------------------
// Kernel 1: pack x[64,8192] f32 -> f16 in the WMMA 16-bit A-operand layout.
// ws layout: [k_tile(256)][m_tile(4)][lane(32)][half(16)]  (1 MB, L2-resident)
// ---------------------------------------------------------------------------
__global__ void pack_a_kernel(const float* __restrict__ x,
                              _Float16* __restrict__ wsA) {
    int id = blockIdx.x * blockDim.x + threadIdx.x;   // 524288 ids
    int h      = id & 15;
    int lane   = (id >> 4) & 31;
    int m_tile = (id >> 9) & 3;
    int k_tile = id >> 11;
    int v = h >> 1, p = h & 1;
    int kb = ((lane & 16) ? 8 : 0) + ((v & 4) ? 16 : 0) + ((v & 3) << 1) + p;
    int m  = m_tile * 16 + (lane & 15);
    int kg = k_tile * QB + kb;
    wsA[id] = (_Float16)x[m * K_DIM + kg];
}

// ---------------------------------------------------------------------------
// Kernel 2: out[m,n] = bias[n]  (gemm accumulates on top via f32 atomics)
// ---------------------------------------------------------------------------
__global__ void bias_init_kernel(const float* __restrict__ bias,
                                 float* __restrict__ out) {
    int id = blockIdx.x * blockDim.x + threadIdx.x;   // 524288 ids
    out[id] = bias[id & (N_DIM - 1)];
}

// ---------------------------------------------------------------------------
// LDS offset (in ints) of K-row r of the staged 32x128 raw tile.
// ---------------------------------------------------------------------------
__device__ __forceinline__ int lds_row(int r) {
    return (r & 16) ? (HI_OFF + (r & 15) * ROW_I) : (r * ROW_I);
}

// Stage one 32x128 int32 weight tile (16 KB) into LDS.
// 1024 16-byte chunks; each of the 256 threads moves 4 chunks.
__device__ __forceinline__ void stage_tile(const int* __restrict__ gbase,
                                           int* lbuf, int tid) {
    #pragma unroll
    for (int j = 0; j < 4; ++j) {
        const int flat = j * 256 + tid;         // chunk id
        const int row  = flat >> 5;             // 32 chunks per 128-int row
        const int c    = (flat & 31) << 2;      // int offset within row
        const int* g = gbase + (size_t)row * N_DIM + c;
        int* l = lbuf + lds_row(row) + c;
#if USE_ASYNC
        __builtin_amdgcn_global_load_async_to_lds_b128(
            (gvs4i*)(uintptr_t)g,
            (lvs4i*)(unsigned)(uintptr_t)l,   // flat addr low 32b == LDS offset
            0, 0);
#else
        *(int4*)l = *(const int4*)g;
#endif
    }
}

// ---------------------------------------------------------------------------
// Kernel 3: block-dequant WMMA GEMM.
//   stage raw(b+1) async -> wait -> barrier ->
//   cooperative dequant of raw(b) into f16 B-operand-layout LDS (once!) ->
//   barrier -> each wave: 1 A load + 4 contiguous B loads + 4 WMMAs.
// grid = (N/128, KSPLIT). 256 threads = 8 wave32.
// ---------------------------------------------------------------------------
__global__ void __launch_bounds__(256)
gemm_kernel(const _Float16* __restrict__ wsA,
            const float* __restrict__ scales,
            const float* __restrict__ zeros,
            const int*   __restrict__ qw,
            float*       __restrict__ out) {
    __shared__ int lbuf[2][BUF_I];
    // dequantized B operands: [tile(8)][lane(32)][half(16)] f16 = 8 KB
    __shared__ alignas(32) _Float16 bstage[NTILES * 32 * 16];

    const int tid     = threadIdx.x;
    const int lane    = tid & 31;
    const int wave    = tid >> 5;
    const int m_tile  = wave & 3;
    const int n_group = wave >> 2;
    const int col     = lane & 15;          // C layout: lane -> column
    const int khalf   = lane & 16;
    const int n_tile0 = blockIdx.x * NT;
    const int b0      = blockIdx.y * KT_PER_WG;

    // dequant-phase role: thread tid produces the v16h consumed by
    // lane (tid&31) of tile (tid>>5): column dq_col, K rows 8h+0..7 then 16+8h+0..7 (h=(tid>>4)&1)
    const int dq_T   = tid >> 5;
    const int dq_l   = tid & 31;
    const int dq_col = dq_T * 16 + (dq_l & 15);
    const int dq_hi  = (dq_l & 16) ? 8 * ROW_I : 0;

    v8f acc[4];
    #pragma unroll
    for (int t = 0; t < 4; ++t)
        #pragma unroll
        for (int i = 0; i < 8; ++i) acc[t][i] = 0.0f;

    const v16h* Av = (const v16h*)wsA;
    v16h* Bv = (v16h*)bstage;

    // prologue: stage quant-block b0 into buffer 0
    stage_tile(qw + (size_t)(b0 * QB) * N_DIM + n_tile0, lbuf[0], tid);

    for (int bt = 0; bt < KT_PER_WG; ++bt) {
        const int b = b0 + bt;

        if (bt + 1 < KT_PER_WG) {
            // overlap: DMA next block while we dequant/compute this one
            stage_tile(qw + (size_t)((b + 1) * QB) * N_DIM + n_tile0,
                       lbuf[(bt + 1) & 1], tid);
#if USE_ASYNC
            __builtin_amdgcn_s_wait_asynccnt(4);  // block b's 4 ops retired
#endif
        } else {
#if USE_ASYNC
            __builtin_amdgcn_s_wait_asynccnt(0);
#endif
        }
        __syncthreads();   // raw(b) staged; consume(b-1) of bstage finished

        // ---- cooperative dequant: each element converted exactly once ----
        {
            const int* dqr = lbuf[bt & 1] + dq_hi + dq_col;
            const float s  = scales[b * N_DIM + n_tile0 + dq_col];
            const float mz = -s * zeros[b * N_DIM + n_tile0 + dq_col];
            v16h bm;
            #pragma unroll
            for (int kk = 0; kk < 16; ++kk)
                bm[kk] = (_Float16)__builtin_fmaf(s, (float)dqr[((kk & 8) ? HI_OFF : 0) + (kk & 7) * ROW_I], mz);
            Bv[tid] = bm;                    // contiguous 32B store
        }
        __syncthreads();   // bstage ready; raw(b) reads done (safe to overwrite)

        // ---- consume: 1 A load + 4 B loads + 4 WMMAs per wave ----
        v16h a = Av[(b * 4 + m_tile) * 32 + lane];
        #pragma unroll
        for (int t = 0; t < 4; ++t) {
            v16h bmat = Bv[(n_group * 4 + t) * 32 + lane];
            acc[t] = __builtin_amdgcn_wmma_f32_16x16x32_f16(
                         false, a, false, bmat, (short)0, acc[t], false, false);
        }
    }

    // C/D layout: vgpr i, lanes 0-15 -> M=i, lanes 16-31 -> M=i+8; N = col
    const int rbase = m_tile * 16 + (khalf ? 8 : 0);
    #pragma unroll
    for (int t = 0; t < 4; ++t) {
        const int n = n_tile0 + n_group * 64 + t * 16 + col;
        #pragma unroll
        for (int i = 0; i < 8; ++i)
            atomicAdd(out + (size_t)(rbase + i) * N_DIM + n, acc[t][i]);
    }
}

// ---------------------------------------------------------------------------
extern "C" void kernel_launch(void* const* d_in, const int* in_sizes, int n_in,
                              void* d_out, int out_size, void* d_ws, size_t ws_size,
                              hipStream_t stream) {
    (void)in_sizes; (void)n_in; (void)out_size; (void)ws_size;
    const float* x      = (const float*)d_in[0];
    const float* scales = (const float*)d_in[1];
    const float* zeros  = (const float*)d_in[2];
    const float* bias   = (const float*)d_in[3];
    const int*   qw     = (const int*)d_in[4];
    float* out = (float*)d_out;
    _Float16* wsA = (_Float16*)d_ws;   // 1 MB packed A

    pack_a_kernel<<<2048, 256, 0, stream>>>(x, wsA);
    bias_init_kernel<<<2048, 256, 0, stream>>>(bias, out);

    dim3 grid(N_DIM / NT, KSPLIT, 1);
    gemm_kernel<<<grid, 256, 0, stream>>>(wsA, scales, zeros, qw, out);
}
